// OmniFractalCore_30709016167038
// MI455X (gfx1250) — matrix operand order repaired and checked by reading, not compiled, not run on hardware
//
#include <hip/hip_runtime.h>
#include <math.h>

// ---------------------------------------------------------------------------
// Types for CDNA5 WMMA (gfx1250, wave32)
// ---------------------------------------------------------------------------
typedef __bf16 bf16;
typedef __attribute__((ext_vector_type(16))) __bf16 bf16x16;
typedef __attribute__((ext_vector_type(8)))  __bf16 bf16x8;
typedef __attribute__((ext_vector_type(8)))  float  f32x8;

union BF16x16U { bf16x16 v; bf16x8 h[2]; };

__device__ __forceinline__ int lane_id() { return (int)(threadIdx.x & 31u); }

__device__ __forceinline__ f32x8 zero8() {
  f32x8 z = {0.f,0.f,0.f,0.f,0.f,0.f,0.f,0.f};
  return z;
}

// ---------------------------------------------------------------------------
// CDNA5 async global->LDS copy (ASYNCcnt-tracked), with sync fallback.
// Builtin parameter type (from clang diagnostic): v4i in AS1 / AS3.
// ---------------------------------------------------------------------------
#if defined(__HIP_DEVICE_COMPILE__) && __has_builtin(__builtin_amdgcn_global_load_async_to_lds_b128)
#define HAVE_ASYNC_LDS 1
typedef int v4i __attribute__((vector_size(16)));
typedef __attribute__((address_space(1))) v4i gv4i_t;
typedef __attribute__((address_space(3))) v4i lv4i_t;
#else
#define HAVE_ASYNC_LDS 0
#endif

__device__ __forceinline__ void async_copy16(bf16* lds_dst, const bf16* gsrc) {
#if HAVE_ASYNC_LDS
  __builtin_amdgcn_global_load_async_to_lds_b128(
      (gv4i_t*)gsrc, (lv4i_t*)lds_dst, 0, 0);
#else
  *reinterpret_cast<bf16x8*>(lds_dst) = *reinterpret_cast<const bf16x8*>(gsrc);
#endif
}

__device__ __forceinline__ void async_wait_all() {
#if HAVE_ASYNC_LDS
#if __has_builtin(__builtin_amdgcn_s_wait_asynccnt)
  __builtin_amdgcn_s_wait_asynccnt(0);
#endif
#endif
}

// A fragment (16x32, bf16): source is row-major [row][k], stride ld (halves).
// Lane l holds row (l&15); elements 0..7 -> K = base + (l<16?0:8) + {0..7},
// elements 8..15 -> same + 16.  (ISA 7.12.2 16-bit A layout.)
__device__ __forceinline__ bf16x16 load_a_frag(const bf16* p, int ld) {
  int l = lane_id();
  const bf16* r = p + (size_t)(l & 15) * ld + ((l >> 4) << 3);
  BF16x16U f;
  f.h[0] = *reinterpret_cast<const bf16x8*>(r);
  f.h[1] = *reinterpret_cast<const bf16x8*>(r + 16);
  return f.v;
}

// B fragment (32x16 K x N, bf16): source stored N-major (i.e. B^T row-major),
// [n][k], stride ld. Lane l holds N=(l&15); elements 0..7 -> K = (l<16?0:8) + {0..7},
// elements 8..15 -> same + 16 (matches A layout).
__device__ __forceinline__ bf16x16 load_b_frag(const bf16* p, int ld) {
  int l = lane_id();
  const bf16* r = p + (size_t)(l & 15) * ld + ((l >> 4) << 3);
  BF16x16U f;
  f.h[0] = *reinterpret_cast<const bf16x8*>(r);
  f.h[1] = *reinterpret_cast<const bf16x8*>(r + 16);
  return f.v;
}

__device__ __forceinline__ f32x8 wmma_bf16(bf16x16 a, bf16x16 b, f32x8 c) {
  return __builtin_amdgcn_wmma_f32_16x16x32_bf16(false, a, false, b,
                                                 (short)0, c, false, false);
}

// ---------------------------------------------------------------------------
// Problem constants
// ---------------------------------------------------------------------------
#define BATCH   2
#define TSEQ    1024
#define DMODEL  512
#define NHEAD   8
#define HDIM    64
#define NLAYER  6
#define NVOCAB  50000
#define MAXDEP  4
#define TDTOT   (BATCH*TSEQ)     // 2048 token rows

// ---------------------------------------------------------------------------
// Small utility kernels
// ---------------------------------------------------------------------------
__global__ void convert_f32_bf16_kernel(const float* __restrict__ a,
                                        bf16* __restrict__ b, int n) {
  int i = blockIdx.x * blockDim.x + threadIdx.x;
  if (i < n) b[i] = (bf16)a[i];
}

// Transposing fp32 -> bf16 weight conversion: in a[z][K][N] -> out bt[z][N][K].
// 32x32 tiles via LDS, 256 threads (32x8).
__global__ __launch_bounds__(256)
void transpose_f32_bf16_kernel(const float* __restrict__ a,
                               bf16* __restrict__ bt, int K, int N) {
  __shared__ float tile[32][33];
  const float* az = a + (size_t)blockIdx.z * K * N;
  bf16* bz = bt + (size_t)blockIdx.z * K * N;
  int nb = blockIdx.x * 32, kb = blockIdx.y * 32;
  int tx = threadIdx.x & 31, ty = threadIdx.x >> 5;
  for (int i = ty; i < 32; i += 8) {
    int k = kb + i, n = nb + tx;
    tile[i][tx] = (k < K && n < N) ? az[(size_t)k * N + n] : 0.f;
  }
  __syncthreads();
  for (int i = ty; i < 32; i += 8) {
    int n = nb + i, k = kb + tx;
    if (n < N && k < K) bz[(size_t)n * K + k] = (bf16)tile[tx][i];
  }
}

__global__ void embed_kernel(const int* __restrict__ ids,
                             const float* __restrict__ emb,
                             float* __restrict__ x, bf16* __restrict__ xb) {
  int row = blockIdx.x;
  int id = ids[row];
  const float* e = emb + (size_t)id * DMODEL;
  for (int c = threadIdx.x; c < DMODEL; c += blockDim.x) {
    float v = e[c];
    x[(size_t)row * DMODEL + c] = v;
    xb[(size_t)row * DMODEL + c] = (bf16)v;
  }
}

__global__ void init_layer_kernel(float* __restrict__ total,
                                  float* __restrict__ active, int n) {
  int i = blockIdx.x * blockDim.x + threadIdx.x;
  if (i < n) total[i] = 0.f;
  if (i == 0) *active = 1.f;
}

__global__ void zero1_kernel(float* p) { if (threadIdx.x == 0) *p = 0.f; }

__global__ void gate_kernel(const float* __restrict__ ent_sum,
                            float* __restrict__ active,
                            float* __restrict__ contrib,
                            int d, float inv_norm) {
  if (threadIdx.x == 0) {
    float a = *active;
    *contrib = a;
    float entn = (*ent_sum) * inv_norm;
    float cond = (entn < 0.8f && d < MAXDEP) ? 1.f : 0.f;
    *active = a * cond;
  }
}

__global__ void accum_kernel(float* __restrict__ total,
                             const float* __restrict__ outf,
                             const float* __restrict__ contrib, int n) {
  float c = *contrib;
  int i = blockIdx.x * blockDim.x + threadIdx.x;
  if (i < n) total[i] += c * outf[i];
}

// ---------------------------------------------------------------------------
// bf16 WMMA GEMM with pre-transposed B:
//   C[M,N] = A[M,K] @ B[K,N] + bias (+residual),  B given N-major as Bt[N][K].
// Block tile 128x128, K-step 32, 256 threads = 8 waves, wave tile 32x64.
// Double-buffered LDS; staging uses GLOBAL_LOAD_ASYNC_TO_LDS_B128 when
// available (ASYNCcnt + s_wait_asynccnt), else sync vector copies.
// ---------------------------------------------------------------------------
#define GBM 128
#define GBN 128
#define GBK 32
#define LDT 40   // padded LDS stride in halves (80B rows, 16B-aligned)

__device__ __forceinline__ void stage_tile(bf16* As, bf16* Bs,
                                           const bf16* __restrict__ A,
                                           const bf16* __restrict__ Bt,
                                           int m_base, int n_base, int kt,
                                           int N, int K, int tid) {
  // A tile 128x32: 512 units of 8 halves (16B), 2 per thread
  for (int u = tid; u < 512; u += 256) {
    int r = u >> 2, c = (u & 3) << 3;
    async_copy16(&As[r * LDT + c], &A[(size_t)(m_base + r) * K + kt + c]);
  }
  // Bt tile 128x32 (N-major rows): clamp OOB rows to a valid one; results for
  // those columns are never stored (epilogue col<N guard).
  for (int u = tid; u < 512; u += 256) {
    int r = u >> 2, c = (u & 3) << 3;
    int n = n_base + r; if (n > N - 1) n = N - 1;
    async_copy16(&Bs[r * LDT + c], &Bt[(size_t)n * K + kt + c]);
  }
}

__global__ __launch_bounds__(256)
void gemm_bf16_kernel(const bf16* __restrict__ A, const bf16* __restrict__ Bt,
                      const float* __restrict__ bias,
                      const float* __restrict__ residual,
                      float* __restrict__ Cf, bf16* __restrict__ Cb,
                      int M, int N, int K) {
  __shared__ __align__(16) bf16 As[2][GBM * LDT];
  __shared__ __align__(16) bf16 Bs[2][GBN * LDT];

  int n_base = blockIdx.x * GBN;
  int m_base = blockIdx.y * GBM;
  int tid = threadIdx.x;
  int wid = tid >> 5;
  int wm = (wid & 3) * 32;      // wave row offset in block tile
  int wn = (wid >> 2) * 64;     // wave col offset in block tile

  f32x8 acc[2][4];
  for (int i = 0; i < 2; i++)
    for (int j = 0; j < 4; j++) acc[i][j] = zero8();

  const int nk = K / GBK;
  stage_tile(As[0], Bs[0], A, Bt, m_base, n_base, 0, N, K, tid);

  for (int it = 0; it < nk; it++) {
    async_wait_all();          // stage `it` async copies landed in LDS
    __syncthreads();           // ...visible block-wide
    if (it + 1 < nk)           // overlap next stage with this stage's math
      stage_tile(As[(it + 1) & 1], Bs[(it + 1) & 1], A, Bt,
                 m_base, n_base, (it + 1) * GBK, N, K, tid);

    const bf16* Ab = As[it & 1];
    const bf16* Bb = Bs[it & 1];
    bf16x16 af[2], bfr[4];
    for (int i = 0; i < 2; i++) af[i] = load_a_frag(&Ab[(wm + i * 16) * LDT], LDT);
    for (int j = 0; j < 4; j++) bfr[j] = load_b_frag(&Bb[(wn + j * 16) * LDT], LDT);
    for (int i = 0; i < 2; i++)
      for (int j = 0; j < 4; j++)
        acc[i][j] = wmma_bf16(af[i], bfr[j], acc[i][j]);
    __syncthreads();           // retire readers before buffer reuse
  }

  // Epilogue: C layout lane holds (row = r + 8*(l>>4), col = l&15)
  int l = lane_id();
  int cn = l & 15, hi = l >> 4;
  for (int i = 0; i < 2; i++)
    for (int j = 0; j < 4; j++) {
      int col = n_base + wn + j * 16 + cn;
      if (col >= N) continue;
      float bv = bias ? bias[col] : 0.f;
      for (int r = 0; r < 8; r++) {
        int row = m_base + wm + i * 16 + r + 8 * hi;
        float v = acc[i][j][r] + bv;
        size_t idx = (size_t)row * N + col;
        if (residual) v += residual[idx];
        if (Cf) Cf[idx] = v;
        if (Cb) Cb[idx] = (bf16)v;
      }
    }
}

// ---------------------------------------------------------------------------
// Flash attention with streaming entropy.
// grid = B*H*16 blocks, 128 threads = 4 waves, each wave owns 16 query rows.
// qkv bf16 [B*T][3*512]; out [B*T][512]; ent_accum += sum over rows of
// (log S1 - S2/S1).
// ---------------------------------------------------------------------------
#define VSTR 40
#define PSTR 40

__global__ __launch_bounds__(128)
void attn_kernel(const bf16* __restrict__ qkv,
                 float* __restrict__ outf, bf16* __restrict__ outb,
                 float* __restrict__ ent_accum) {
  __shared__ __align__(16) bf16 Vt[HDIM * VSTR];        // V^T chunk [feat][key]
  __shared__ __align__(16) bf16 Pl[4][16 * PSTR];       // per-wave P tile

  int blk = blockIdx.x;
  int qt = blk & 15;            // 16 query tiles of 64 rows
  int h  = (blk >> 4) & 7;
  int b  = blk >> 7;
  int tid = threadIdx.x;
  int wid = tid >> 5;
  int l = tid & 31;
  int cn = l & 15, hi = l >> 4;
  int row0 = qt * 64 + wid * 16;          // wave's query tile base (in seq)
  const int ldq = 3 * DMODEL;
  const float scale = 0.044194173824159216f;   // 512^-0.5

  // Q fragments (feature dim 64 -> two K=32 chunks), loaded direct from global
  const bf16* qbase = qkv + (size_t)(b * TSEQ + row0) * ldq + h * HDIM;
  bf16x16 qf[2];
  qf[0] = load_a_frag(qbase, ldq);
  qf[1] = load_a_frag(qbase + 32, ldq);

  float rowM[8], s1[8], s2[8];
  f32x8 o[4];
  for (int r = 0; r < 8; r++) { rowM[r] = -1e30f; s1[r] = 0.f; s2[r] = 0.f; }
  for (int j = 0; j < 4; j++) o[j] = zero8();

  for (int kb = 0; kb < TSEQ; kb += 32) {
    __syncthreads();   // Vt reuse guard
    // cooperative V^T staging: V chunk [32 key][64 feat] -> Vt[feat][key]
    for (int u = tid; u < 256; u += 128) {
      int key = u >> 3, f0 = (u & 7) << 3;
      bf16 tmp[8];
      *reinterpret_cast<bf16x8*>(tmp) = *reinterpret_cast<const bf16x8*>(
          &qkv[(size_t)(b * TSEQ + kb + key) * ldq + 2 * DMODEL + h * HDIM + f0]);
      for (int j = 0; j < 8; j++) Vt[(f0 + j) * VSTR + key] = tmp[j];
    }

    // scores for 32 keys = two 16-key sub-blocks; K (row-major) is already B^T
    f32x8 sc[2];
    for (int s = 0; s < 2; s++) {
      const bf16* kbase =
          qkv + (size_t)(b * TSEQ + kb + s * 16) * ldq + DMODEL + h * HDIM;
      bf16x16 kf0 = load_b_frag(kbase, ldq);
      bf16x16 kf1 = load_b_frag(kbase + 32, ldq);
      f32x8 z = zero8();
      z = wmma_bf16(qf[0], kf0, z);
      z = wmma_bf16(qf[1], kf1, z);
      sc[s] = z * scale;
    }

    // online softmax + entropy bookkeeping (per-lane partial S1/S2)
    float alpha[8];
    for (int r = 0; r < 8; r++) {
      float m = fmaxf(sc[0][r], sc[1][r]);
      m = fmaxf(m, __shfl_xor(m, 1, 32));
      m = fmaxf(m, __shfl_xor(m, 2, 32));
      m = fmaxf(m, __shfl_xor(m, 4, 32));
      m = fmaxf(m, __shfl_xor(m, 8, 32));
      float nm = fmaxf(rowM[r], m);
      float d = rowM[r] - nm;
      float a = __expf(d);
      s2[r] = a * (s2[r] + d * s1[r]);
      s1[r] = a * s1[r];
      rowM[r] = nm;
      alpha[r] = a;
      float d0 = sc[0][r] - nm, d1 = sc[1][r] - nm;
      float e0 = __expf(d0), e1 = __expf(d1);
      s1[r] += e0 + e1;
      s2[r] += e0 * d0 + e1 * d1;
      Pl[wid][(r + 8 * hi) * PSTR + 0 * 16 + cn] = (bf16)e0;
      Pl[wid][(r + 8 * hi) * PSTR + 1 * 16 + cn] = (bf16)e1;
    }
    // rescale running output by alpha (row-matched layout)
    for (int j = 0; j < 4; j++)
      for (int r = 0; r < 8; r++) o[j][r] *= alpha[r];

    __syncthreads();   // Vt + P visible
    bf16x16 pf = load_a_frag(&Pl[wid][0], PSTR);
    for (int j = 0; j < 4; j++) {
      bf16x16 vf = load_b_frag(&Vt[(j * 16) * VSTR], VSTR);
      o[j] = wmma_bf16(pf, vf, o[j]);
    }
  }

  // finalize: cross-lane reduce S1/S2 over the 16 lanes sharing each row
  for (int r = 0; r < 8; r++) {
    s1[r] += __shfl_xor(s1[r], 1, 32);
    s1[r] += __shfl_xor(s1[r], 2, 32);
    s1[r] += __shfl_xor(s1[r], 4, 32);
    s1[r] += __shfl_xor(s1[r], 8, 32);
    s2[r] += __shfl_xor(s2[r], 1, 32);
    s2[r] += __shfl_xor(s2[r], 2, 32);
    s2[r] += __shfl_xor(s2[r], 4, 32);
    s2[r] += __shfl_xor(s2[r], 8, 32);
  }
  if (cn == 0) {   // lanes 0 (rows 0..7) and 16 (rows 8..15)
    float es = 0.f;
    for (int r = 0; r < 8; r++)
      es += __logf(s1[r]) - s2[r] / s1[r];
    atomicAdd(ent_accum, es);
  }
  // normalize and store out[b,t, h*64 + feat]
  for (int j = 0; j < 4; j++)
    for (int r = 0; r < 8; r++) {
      float v = o[j][r] / s1[r];
      int orow = b * TSEQ + row0 + r + 8 * hi;
      int col = h * HDIM + j * 16 + cn;
      size_t idx = (size_t)orow * DMODEL + col;
      outf[idx] = v;
      outb[idx] = (bf16)v;
    }
}

// ---------------------------------------------------------------------------
// LayerNorm over D=512, one block per token row; writes bf16 (GEMM A input)
// ---------------------------------------------------------------------------
__global__ __launch_bounds__(256)
void ln_kernel(const float* __restrict__ x, const float* __restrict__ g,
               const float* __restrict__ be, bf16* __restrict__ out) {
  __shared__ float red[256];
  int row = blockIdx.x;
  const float* xr = x + (size_t)row * DMODEL;
  float s = 0.f;
  for (int c = threadIdx.x; c < DMODEL; c += 256) s += xr[c];
  red[threadIdx.x] = s; __syncthreads();
  for (int st = 128; st > 0; st >>= 1) {
    if (threadIdx.x < st) red[threadIdx.x] += red[threadIdx.x + st];
    __syncthreads();
  }
  float mu = red[0] * (1.f / DMODEL);
  __syncthreads();
  float v = 0.f;
  for (int c = threadIdx.x; c < DMODEL; c += 256) {
    float d = xr[c] - mu; v += d * d;
  }
  red[threadIdx.x] = v; __syncthreads();
  for (int st = 128; st > 0; st >>= 1) {
    if (threadIdx.x < st) red[threadIdx.x] += red[threadIdx.x + st];
    __syncthreads();
  }
  float rs = rsqrtf(red[0] * (1.f / DMODEL) + 1e-5f);
  for (int c = threadIdx.x; c < DMODEL; c += 256)
    out[(size_t)row * DMODEL + c] = (bf16)((xr[c] - mu) * rs * g[c] + be[c]);
}

// ---------------------------------------------------------------------------
// Host orchestration
// ---------------------------------------------------------------------------
extern "C" void kernel_launch(void* const* d_in, const int* in_sizes, int n_in,
                              void* d_out, int out_size, void* d_ws, size_t ws_size,
                              hipStream_t stream) {
  (void)in_sizes; (void)n_in; (void)out_size; (void)ws_size;
  const int*   ids  = (const int*)  d_in[0];
  const float* emb  = (const float*)d_in[1];
  const float* Wqkv = (const float*)d_in[2];
  const float* bqkv = (const float*)d_in[3];
  const float* Wout = (const float*)d_in[4];
  const float* bout = (const float*)d_in[5];
  const float* gam  = (const float*)d_in[6];
  const float* bet  = (const float*)d_in[7];
  const float* Wv   = (const float*)d_in[8];
  const float* bv   = (const float*)d_in[9];
  float* out = (float*)d_out;

  // workspace carving
  char* w = (char*)d_ws;
  size_t off = 0;
  auto alloc = [&](size_t bytes) -> void* {
    off = (off + 255) & ~(size_t)255;
    void* p = w + off;
    off += bytes;
    return p;
  };
  const size_t NX = (size_t)TDTOT * DMODEL;           // 1M elems
  float* x_a      = (float*)alloc(NX * 4);
  float* x_b      = (float*)alloc(NX * 4);
  bf16*  xb_a     = (bf16*) alloc(NX * 2);
  bf16*  xb_b     = (bf16*) alloc(NX * 2);
  float* total    = (float*)alloc(NX * 4);
  bf16*  total_bf = (bf16*) alloc(NX * 2);
  bf16*  lvl_a    = (bf16*) alloc(NX * 2);
  bf16*  lvl_b    = (bf16*) alloc(NX * 2);
  float* out_f    = (float*)alloc(NX * 4);
  bf16*  qkv_bf   = (bf16*) alloc((size_t)TDTOT * 3 * DMODEL * 2);
  bf16*  xn_bf    = (bf16*) alloc(NX * 2);
  bf16*  wqkvT    = (bf16*) alloc((size_t)NLAYER * DMODEL * 3 * DMODEL * 2);  // [L][3D][D]
  bf16*  woutT    = (bf16*) alloc((size_t)NLAYER * DMODEL * DMODEL * 2);      // [L][D][D]
  bf16*  wvT      = (bf16*) alloc((size_t)DMODEL * NVOCAB * 2);               // [V][D]
  float* scal     = (float*)alloc(64);
  float* ent_sum  = scal + 0;
  float* active   = scal + 1;
  float* contrib  = scal + 2;

  // one-time transposing weight conversion fp32 -> bf16 (B becomes N-major)
  transpose_f32_bf16_kernel<<<dim3(48, 16, NLAYER), 256, 0, stream>>>(
      Wqkv, wqkvT, DMODEL, 3 * DMODEL);
  transpose_f32_bf16_kernel<<<dim3(16, 16, NLAYER), 256, 0, stream>>>(
      Wout, woutT, DMODEL, DMODEL);
  transpose_f32_bf16_kernel<<<dim3((NVOCAB + 31) / 32, 16, 1), 256, 0, stream>>>(
      Wv, wvT, DMODEL, NVOCAB);

  // embedding gather
  embed_kernel<<<TDTOT, 256, 0, stream>>>(ids, emb, x_a, xb_a);

  const float inv_norm = 1.f / ((float)(BATCH * NHEAD * TSEQ) * logf((float)TSEQ));
  const int nElem = (int)NX;

  float* xf = x_a;  float* xf2 = x_b;
  bf16*  xb = xb_a; bf16*  xb2 = xb_b;

  for (int layer = 0; layer < NLAYER; layer++) {
    init_layer_kernel<<<(nElem + 255) / 256, 256, 0, stream>>>(total, active, nElem);

    bf16* cur = xb;
    bf16* pa = lvl_a;
    bf16* pb = lvl_b;
    for (int d = 0; d < MAXDEP + 1; d++) {
      zero1_kernel<<<1, 32, 0, stream>>>(ent_sum);
      // qkv = cur @ Wqkv[l] + bqkv[l]   (bf16 out only)
      gemm_bf16_kernel<<<dim3(12, 16), 256, 0, stream>>>(
          cur, wqkvT + (size_t)layer * DMODEL * 3 * DMODEL,
          bqkv + (size_t)layer * 3 * DMODEL, nullptr,
          nullptr, qkv_bf, TDTOT, 3 * DMODEL, DMODEL);
      // attention level
      attn_kernel<<<BATCH * NHEAD * 16, 128, 0, stream>>>(qkv_bf, out_f, pa, ent_sum);
      // entropy gate: contrib = active;  active *= gate(ent, d)
      gate_kernel<<<1, 32, 0, stream>>>(ent_sum, active, contrib, d, inv_norm);
      // total += contrib * out
      accum_kernel<<<(nElem + 255) / 256, 256, 0, stream>>>(total, out_f, contrib, nElem);
      cur = pa;
      bf16* t = pa; pa = pb; pb = t;
    }

    // x = total @ Wout[l] + bout[l] + x
    convert_f32_bf16_kernel<<<(nElem + 255) / 256, 256, 0, stream>>>(total, total_bf, nElem);
    gemm_bf16_kernel<<<dim3(4, 16), 256, 0, stream>>>(
        total_bf, woutT + (size_t)layer * DMODEL * DMODEL,
        bout + (size_t)layer * DMODEL, xf,
        xf2, xb2, TDTOT, DMODEL, DMODEL);
    { float* tf = xf; xf = xf2; xf2 = tf; }
    { bf16* tb = xb; xb = xb2; xb2 = tb; }
  }

  // final layernorm + vocab projection
  ln_kernel<<<TDTOT, 256, 0, stream>>>(xf, gam, bet, xn_bf);
  gemm_bf16_kernel<<<dim3((NVOCAB + GBN - 1) / GBN, TDTOT / GBM), 256, 0, stream>>>(
      xn_bf, wvT, bv, nullptr, out, nullptr, TDTOT, NVOCAB, DMODEL);
}
